// MultiHeadAttention_19164144075402
// MI455X (gfx1250) — matrix operand order repaired and checked by reading, not compiled, not run on hardware
//
#include <hip/hip_runtime.h>

// ---------------------------------------------------------------------------
// CDNA5 / gfx1250 fused multi-head attention (wave32, v_wmma_f32_16x16x32_bf16)
// with async global->LDS staging (ASYNCcnt) and global_prefetch_b8 lookahead.
// ---------------------------------------------------------------------------

typedef __bf16          v16bf __attribute__((ext_vector_type(16)));
typedef float           v8f   __attribute__((ext_vector_type(8)));
typedef float           v4f   __attribute__((ext_vector_type(4)));
typedef int             v4i   __attribute__((ext_vector_type(4)));
typedef unsigned short  v4u   __attribute__((ext_vector_type(4)));
typedef unsigned short  v8u   __attribute__((ext_vector_type(8)));
typedef unsigned short  v16u  __attribute__((ext_vector_type(16)));

static constexpr int BB  = 4;     // batch
static constexpr int SS  = 2048;  // sequence
static constexpr int DD  = 1024;  // model dim
static constexpr int HH  = 16;    // heads
static constexpr int DPH = 64;    // depth per head

#if defined(__has_builtin)
#  if __has_builtin(__builtin_amdgcn_global_load_async_to_lds_b128)
#    define HAVE_ASYNC_LDS 1
#  endif
#endif
#ifndef HAVE_ASYNC_LDS
#  define HAVE_ASYNC_LDS 0
#endif

typedef __attribute__((address_space(1))) v4i* glob_v4i_ptr;
typedef __attribute__((address_space(3))) v4i* lds_v4i_ptr;

__device__ __forceinline__ unsigned short f2bf(float f) {
  unsigned int u = __builtin_bit_cast(unsigned int, f);
  u += 0x7fffu + ((u >> 16) & 1u);           // round-to-nearest-even
  return (unsigned short)(u >> 16);
}

// 16-byte global -> LDS copy: async DMA path (ASYNCcnt) or manual fallback.
__device__ __forceinline__ void copy_b128_g2l(const unsigned short* g,
                                              unsigned short* l) {
#if HAVE_ASYNC_LDS
  __builtin_amdgcn_global_load_async_to_lds_b128((glob_v4i_ptr)g, (lds_v4i_ptr)l,
                                                 0, 0);
#else
  *(v8u*)l = *(const v8u*)g;
#endif
}
__device__ __forceinline__ void async_wait() {
#if HAVE_ASYNC_LDS
  asm volatile("s_wait_asynccnt 0x0" ::: "memory");
#endif
}

__device__ __forceinline__ v8f wmma_bf16(v16bf a, v16bf b, v8f c) {
  // (neg_a, A, neg_b, B, c_mod, C, reuse_a, reuse_b)
  return __builtin_amdgcn_wmma_f32_16x16x32_bf16(false, a, false, b,
                                                 (short)0, c, false, false);
}

// A-matrix 16x32 bf16 fragment from LDS laid out row-major [row][k].
__device__ __forceinline__ v16bf load_a_frag(const unsigned short* s, int stride,
                                             int r0, int k0, int lane) {
  const int m  = r0 + (lane & 15);
  const int kb = k0 + ((lane >> 4) << 3);
  v8u lo = *(const v8u*)(s + m * stride + kb);
  v8u hi = *(const v8u*)(s + m * stride + kb + 16);
  v16u t;
#pragma unroll
  for (int i = 0; i < 8; ++i) { t[i] = lo[i]; t[i + 8] = hi[i]; }
  return __builtin_bit_cast(v16bf, t);
}

// B-matrix 32x16 bf16 fragment from LDS holding B-transposed [n][k].
__device__ __forceinline__ v16bf load_b_frag(const unsigned short* s, int stride,
                                             int n0, int k0, int lane) {
  const int n  = n0 + (lane & 15);
  const int kb = k0 + ((lane >> 4) << 3);
  v8u lo = *(const v8u*)(s + n * stride + kb);
  v8u hi = *(const v8u*)(s + n * stride + kb + 16);
  v16u t;
#pragma unroll
  for (int i = 0; i < 8; ++i) { t[i] = lo[i]; t[i + 8] = hi[i]; }
  return __builtin_bit_cast(v16bf, t);
}

// ---------------------------------------------------------------------------
// Generic GEMM:  out[M,N] = A[M,K] @ W[K,N] + bias[N]
// Block tile 128x128, K-step 64, 256 threads = 8 waves arranged 4(M) x 2(N),
// each wave computes 32x64 via 2x4 WMMA accumulators (16 WMMA per stage).
// ---------------------------------------------------------------------------
template <bool A_BF16, bool OUT_BF16>
__global__ __launch_bounds__(256) void wmma_gemm_bias(
    const void* __restrict__ A_, const float* __restrict__ W,
    const float* __restrict__ bias, void* __restrict__ out_,
    int M, int N, int K) {
  __shared__ __align__(16) unsigned short sA[128 * 72];  // [row][k], stride 72
  __shared__ __align__(16) unsigned short sB[128 * 72];  // [n][k]  , stride 72

  const int t = threadIdx.x, lane = t & 31, wave = t >> 5;
  const int bm = blockIdx.x * 128, bn = blockIdx.y * 128;
  const int wm = (wave >> 1) * 32;
  const int wn = (wave & 1) * 64;

  v8f acc[2][4] = {};

  for (int k0 = 0; k0 < K; k0 += 64) {
    // ---- stage A tile (128x64) to LDS as bf16, 16B granularity ----
    if (A_BF16) {
#pragma unroll
      for (int i = 0; i < 4; ++i) {           // 1024 x b128 chunks
        int e = t + i * 256;
        int r = e >> 3, kk0 = (e & 7) * 8;
        copy_b128_g2l((const unsigned short*)A_ + (size_t)(bm + r) * K + k0 + kk0,
                      sA + r * 72 + kk0);
      }
    } else {
#pragma unroll
      for (int i = 0; i < 8; ++i) {           // 2048 x float4 chunks
        int e = t + i * 256;
        int r = e >> 4, kk0 = (e & 15) * 4;
        v4f f = *(const v4f*)((const float*)A_ + (size_t)(bm + r) * K + k0 + kk0);
        v4u u;
#pragma unroll
        for (int j = 0; j < 4; ++j) u[j] = f2bf(f[j]);
        *(v4u*)(sA + r * 72 + kk0) = u;
      }
    }
    // ---- stage W tile (64x128) to LDS transposed [n][k] ----
#pragma unroll
    for (int i = 0; i < 8; ++i) {
      int e = t + i * 256;
      int n0 = (e & 31) * 4, kk = e >> 5;
      v4f f = *(const v4f*)(W + (size_t)(k0 + kk) * N + bn + n0);
#pragma unroll
      for (int j = 0; j < 4; ++j) sB[(n0 + j) * 72 + kk] = f2bf(f[j]);
    }
    // ---- prefetch next tiles while this one computes ----
    if (k0 + 64 < K) {
      int r = t & 127;
      if (A_BF16)
        __builtin_prefetch((const unsigned short*)A_ + (size_t)(bm + r) * K + k0 + 64, 0, 1);
      else
        __builtin_prefetch((const float*)A_ + (size_t)(bm + r) * K + k0 + 64, 0, 1);
      __builtin_prefetch(W + (size_t)(k0 + 64 + (t & 63)) * N + bn + (t >> 6) * 32, 0, 1);
    }
    async_wait();
    __syncthreads();

#pragma unroll
    for (int ks = 0; ks < 2; ++ks) {
      v16bf af[2], bf[4];
#pragma unroll
      for (int mu = 0; mu < 2; ++mu) af[mu] = load_a_frag(sA, 72, wm + mu * 16, ks * 32, lane);
#pragma unroll
      for (int nu = 0; nu < 4; ++nu) bf[nu] = load_b_frag(sB, 72, wn + nu * 16, ks * 32, lane);
#pragma unroll
      for (int mu = 0; mu < 2; ++mu)
#pragma unroll
        for (int nu = 0; nu < 4; ++nu)
          acc[mu][nu] = wmma_bf16(af[mu], bf[nu], acc[mu][nu]);
    }
    __syncthreads();
  }

  // ---- epilogue: C/D layout (VGPR r -> M = r + 8*(lane>>4), N = lane&15) ----
#pragma unroll
  for (int mu = 0; mu < 2; ++mu)
#pragma unroll
    for (int nu = 0; nu < 4; ++nu) {
      const int n = bn + wn + nu * 16 + (lane & 15);
      const float bv = bias[n];
#pragma unroll
      for (int r = 0; r < 8; ++r) {
        const int m = bm + wm + mu * 16 + r + ((lane >> 4) << 3);
        const float v = acc[mu][nu][r] + bv;
        if (OUT_BF16)
          ((unsigned short*)out_)[(size_t)m * N + n] = f2bf(v);
        else
          ((float*)out_)[(size_t)m * N + n] = v;
      }
    }
}

// ---------------------------------------------------------------------------
// Fused attention: per (b, h, 64-row q-tile), two passes over K/V tiles of 128.
// Pass 1: logits via WMMA -> online row max/sum.  Pass 2: recompute logits,
// normalize, *mask*, *group_prob*, stream attn (non-temporal), ctx = P @ V.
// ---------------------------------------------------------------------------
__global__ __launch_bounds__(256) void attn_kernel(
    const unsigned short* __restrict__ qp, const unsigned short* __restrict__ kp,
    const unsigned short* __restrict__ vp, const float* __restrict__ mask,
    const float* __restrict__ gp, float* __restrict__ attn_out,
    unsigned short* __restrict__ ctx) {
  __shared__ __align__(16) unsigned short sQ[64 * 72];    // [m][d]     stride 72
  __shared__ __align__(16) unsigned short sK[128 * 72];   // [kpos][d]  stride 72
  __shared__ __align__(16) unsigned short sV[64 * 136];   // V^T [d][kpos] str 136
  __shared__ __align__(16) unsigned short sP[64 * 136];   // P [m][kpos]   str 136
  __shared__ float pmax[2 * 64], psum[2 * 64];
  __shared__ float rmax[64], rinv[64];

  const int t = threadIdx.x, lane = t & 31, wave = t >> 5;
  const int q0 = blockIdx.x * 64;
  const int h = blockIdx.y, b = blockIdx.z;
  const float scale = 0.125f;  // 1/sqrt(64)

  // ---- load Q tile (64 x 64) once, async (covered by first async_wait) ----
#pragma unroll
  for (int i = 0; i < 2; ++i) {
    int e = t + i * 256;
    int m = e >> 3, dpt0 = (e & 7) * 8;
    copy_b128_g2l(qp + ((size_t)b * SS + q0 + m) * DD + h * DPH + dpt0,
                  sQ + m * 72 + dpt0);
  }

  const int wm = (wave >> 1) * 16;   // logits: wave covers 16 q-rows x 64 k-cols
  const int wn = (wave & 1) * 64;

  float m8[8], s8[8];
#pragma unroll
  for (int r = 0; r < 8; ++r) { m8[r] = -3.0e38f; s8[r] = 0.f; }

  // ======================= pass 1: row max / sum =======================
  for (int kt = 0; kt < SS / 128; ++kt) {
    const int k0 = kt * 128;
    __syncthreads();
#pragma unroll
    for (int i = 0; i < 4; ++i) {             // 1024 x b128 chunks
      int e = t + i * 256;
      int kr = e >> 3, dpt0 = (e & 7) * 8;
      copy_b128_g2l(kp + ((size_t)b * SS + k0 + kr) * DD + h * DPH + dpt0,
                    sK + kr * 72 + dpt0);
    }
    if (kt + 1 < SS / 128)                    // prefetch next K tile
      __builtin_prefetch(kp + ((size_t)b * SS + k0 + 128 + (t & 127)) * DD +
                             h * DPH + (t >> 7) * 32, 0, 1);
    async_wait();
    __syncthreads();

    v8f acc[4] = {};
#pragma unroll
    for (int ks = 0; ks < 2; ++ks) {
      v16bf aq = load_a_frag(sQ, 72, wm, ks * 32, lane);
#pragma unroll
      for (int nu = 0; nu < 4; ++nu) {
        v16bf bk = load_b_frag(sK, 72, wn + nu * 16, ks * 32, lane);
        acc[nu] = wmma_bf16(aq, bk, acc[nu]);
      }
    }
    float mk[4];
#pragma unroll
    for (int nu = 0; nu < 4; ++nu)
      mk[nu] = mask[(size_t)b * SS + k0 + wn + nu * 16 + (lane & 15)] * -1.0e9f;
#pragma unroll
    for (int r = 0; r < 8; ++r) {
      float x[4], lm = -3.0e38f;
#pragma unroll
      for (int nu = 0; nu < 4; ++nu) {
        x[nu] = acc[nu][r] * scale + mk[nu];
        lm = fmaxf(lm, x[nu]);
      }
      const float nm = fmaxf(m8[r], lm);
      float ss = 0.f;
#pragma unroll
      for (int nu = 0; nu < 4; ++nu) ss += __expf(x[nu] - nm);
      s8[r] = s8[r] * __expf(m8[r] - nm) + ss;
      m8[r] = nm;
    }
  }

  // reduce across the 16 lanes sharing each row (butterfly stays in half-wave)
#pragma unroll
  for (int r = 0; r < 8; ++r) {
    float m = m8[r], s = s8[r];
    for (int off = 1; off < 16; off <<= 1) {
      float om = __shfl_xor(m, off, 32);
      float os = __shfl_xor(s, off, 32);
      float nm = fmaxf(m, om);
      s = s * __expf(m - nm) + os * __expf(om - nm);
      m = nm;
    }
    m8[r] = m; s8[r] = s;
  }
  const int wi = wave & 1;
  if ((lane & 15) == 0) {
    const int half = lane >> 4;
#pragma unroll
    for (int r = 0; r < 8; ++r) {
      pmax[wi * 64 + wm + half * 8 + r] = m8[r];
      psum[wi * 64 + wm + half * 8 + r] = s8[r];
    }
  }
  __syncthreads();
  if (t < 64) {
    const float a = pmax[t], c = pmax[64 + t];
    const float nm = fmaxf(a, c);
    const float s = psum[t] * __expf(a - nm) + psum[64 + t] * __expf(c - nm);
    rmax[t] = nm;
    rinv[t] = 1.0f / s;
  }

  const int cm = (wave >> 1) * 16;   // ctx: wave covers 16 q-rows x 32 depth
  const int cn = (wave & 1) * 32;
  v8f cacc[2] = {};

  // ============== pass 2: normalize, emit attn, ctx = P @ V ==============
  for (int kt = 0; kt < SS / 128; ++kt) {
    const int k0 = kt * 128;
    __syncthreads();
#pragma unroll
    for (int i = 0; i < 4; ++i) {             // K tile: async straight copy
      int e = t + i * 256;
      int kr = e >> 3, dpt0 = (e & 7) * 8;
      copy_b128_g2l(kp + ((size_t)b * SS + k0 + kr) * DD + h * DPH + dpt0,
                    sK + kr * 72 + dpt0);
    }
#pragma unroll
    for (int i = 0; i < 4; ++i) {             // V tile: b128 load, transposed store
      int e = t + i * 256;
      int kr = e >> 3, dpt0 = (e & 7) * 8;
      v8u vv = *(const v8u*)(vp + ((size_t)b * SS + k0 + kr) * DD + h * DPH + dpt0);
#pragma unroll
      for (int j = 0; j < 8; ++j) sV[(dpt0 + j) * 136 + kr] = vv[j];
    }
    if (kt + 1 < SS / 128) {
      __builtin_prefetch(kp + ((size_t)b * SS + k0 + 128 + (t & 127)) * DD +
                             h * DPH + (t >> 7) * 32, 0, 1);
      __builtin_prefetch(vp + ((size_t)b * SS + k0 + 128 + (t & 127)) * DD +
                             h * DPH + (t >> 7) * 32, 0, 1);
    }
    async_wait();
    __syncthreads();

    v8f acc[4] = {};
#pragma unroll
    for (int ks = 0; ks < 2; ++ks) {
      v16bf aq = load_a_frag(sQ, 72, wm, ks * 32, lane);
#pragma unroll
      for (int nu = 0; nu < 4; ++nu) {
        v16bf bk = load_b_frag(sK, 72, wn + nu * 16, ks * 32, lane);
        acc[nu] = wmma_bf16(aq, bk, acc[nu]);
      }
    }
    float mk[4];
#pragma unroll
    for (int nu = 0; nu < 4; ++nu)
      mk[nu] = mask[(size_t)b * SS + k0 + wn + nu * 16 + (lane & 15)] * -1.0e9f;
#pragma unroll
    for (int r = 0; r < 8; ++r) {
      const int m = wm + r + ((lane >> 4) << 3);
      const size_t mg = q0 + m;
      const float rm = rmax[m], ri = rinv[m];
      const float* gprow = gp + ((size_t)b * SS + mg) * SS + k0;
      float* arow = attn_out + (((size_t)b * HH + h) * SS + mg) * SS + k0;
#pragma unroll
      for (int nu = 0; nu < 4; ++nu) {
        const int n = wn + nu * 16 + (lane & 15);
        const float x = acc[nu][r] * scale + mk[nu];
        const float p = __expf(x - rm) * ri * __builtin_nontemporal_load(gprow + n);
        __builtin_nontemporal_store(p, arow + n);  // 1.07 GB write-once stream
        sP[m * 136 + n] = f2bf(p);
      }
    }
    __syncthreads();  // P complete before ctx WMMA

#pragma unroll
    for (int ks = 0; ks < 4; ++ks) {
      v16bf ap = load_a_frag(sP, 136, cm, ks * 32, lane);
#pragma unroll
      for (int j = 0; j < 2; ++j) {
        v16bf bv = load_b_frag(sV, 136, cn + j * 16, ks * 32, lane);
        cacc[j] = wmma_bf16(ap, bv, cacc[j]);
      }
    }
  }

  // ---- ctx epilogue -> bf16 workspace in [B,S,H,depth] (= concat) layout ----
#pragma unroll
  for (int j = 0; j < 2; ++j) {
    const int n = cn + j * 16 + (lane & 15);
#pragma unroll
    for (int r = 0; r < 8; ++r) {
      const int m = cm + r + ((lane >> 4) << 3);
      ctx[((size_t)b * SS + q0 + m) * DD + h * DPH + n] = f2bf(cacc[j][r]);
    }
  }
}

// ---------------------------------------------------------------------------
extern "C" void kernel_launch(void* const* d_in, const int* in_sizes, int n_in,
                              void* d_out, int out_size, void* d_ws, size_t ws_size,
                              hipStream_t stream) {
  (void)in_sizes; (void)n_in; (void)out_size; (void)ws_size;
  const float* q    = (const float*)d_in[0];
  const float* k    = (const float*)d_in[1];
  const float* v    = (const float*)d_in[2];
  const float* mask = (const float*)d_in[3];
  const float* gp   = (const float*)d_in[4];
  const float* wq_k = (const float*)d_in[5];
  const float* wq_b = (const float*)d_in[6];
  const float* wk_k = (const float*)d_in[7];
  const float* wk_b = (const float*)d_in[8];
  const float* wv_k = (const float*)d_in[9];
  const float* wv_b = (const float*)d_in[10];
  const float* wo_k = (const float*)d_in[11];
  const float* wo_b = (const float*)d_in[12];

  float* out  = (float*)d_out;                         // [B,S,D]
  float* attn = out + (size_t)BB * SS * DD;            // [B,H,S,S]

  const size_t elems = (size_t)BB * SS * DD;           // per-tensor bf16 ws
  unsigned short* qp  = (unsigned short*)d_ws;
  unsigned short* kp  = qp + elems;
  unsigned short* vp  = kp + elems;
  unsigned short* ctx = vp + elems;

  const int M = BB * SS;  // 8192
  dim3 gg(M / 128, DD / 128);  // 64 x 8
  wmma_gemm_bias<false, true><<<gg, 256, 0, stream>>>(q, wq_k, wq_b, qp, M, DD, DD);
  wmma_gemm_bias<false, true><<<gg, 256, 0, stream>>>(k, wk_k, wk_b, kp, M, DD, DD);
  wmma_gemm_bias<false, true><<<gg, 256, 0, stream>>>(v, wv_k, wv_b, vp, M, DD, DD);

  dim3 ga(SS / 64, HH, BB);  // 32 x 16 x 4
  attn_kernel<<<ga, 256, 0, stream>>>(qp, kp, vp, mask, gp, attn, ctx);

  wmma_gemm_bias<true, false><<<gg, 256, 0, stream>>>(ctx, wo_k, wo_b, out, M, DD, DD);
}
